// NanoDeepSeek_44590350467671
// MI455X (gfx1250) — matrix operand order repaired and checked by reading, not compiled, not run on hardware
//
#include <hip/hip_runtime.h>
#include <math.h>

// ---------------------------------------------------------------------------
// NanoDeepSeek forward on gfx1250 (CDNA5): bf16 WMMA GEMMs everywhere,
// async-to-LDS double-buffered tiles, top-3 gathered MoE, exact GELU,
// causal softmax.
// ---------------------------------------------------------------------------

typedef __attribute__((ext_vector_type(16))) __bf16 v16bf;
typedef __attribute__((ext_vector_type(8)))  __bf16 v8bf;
typedef __attribute__((ext_vector_type(8)))  float  v8f;

#define BTOK   4096          // B*T
#define TSEQ   1024
#define HDIM   1024
#define NHEAD  16
#define DHEAD  64
#define EDIM   512
#define NEXP   10
#define KTOP   3
#define VOCAB  32000
#define NLAYER 4
#define CAP    4096          // per-expert token capacity

// ---- CDNA5 async global->LDS support (guarded; falls back to sync path) ----
#if defined(__has_builtin)
#  if __has_builtin(__builtin_amdgcn_global_load_async_to_lds_b128)
#    define HAVE_ASYNC_LDS 1
#  endif
#  if __has_builtin(__builtin_amdgcn_s_wait_asynccnt)
#    define HAVE_WAIT_ASYNC 1
#  endif
#endif
#ifndef HAVE_ASYNC_LDS
#  define HAVE_ASYNC_LDS 0
#endif
#ifndef HAVE_WAIT_ASYNC
#  define HAVE_WAIT_ASYNC 0
#endif

#if HAVE_ASYNC_LDS
// Builtin signature (from compiler diagnostics):
//   (int4 addrspace(1)* gsrc, int4 addrspace(3)* ldsdst, imm offset, imm cpol)
typedef int async_v4i __attribute__((vector_size(16)));
typedef __attribute__((address_space(1))) async_v4i* async_gptr;
typedef __attribute__((address_space(3))) async_v4i* async_lptr;

__device__ __forceinline__ void wait_async0() {
#if HAVE_WAIT_ASYNC
  __builtin_amdgcn_s_wait_asynccnt(0);
#else
  asm volatile("s_wait_asynccnt 0x0" ::: "memory");
#endif
}
__device__ __forceinline__ void wait_async4() {
#if HAVE_WAIT_ASYNC
  __builtin_amdgcn_s_wait_asynccnt(4);
#else
  asm volatile("s_wait_asynccnt 0x4" ::: "memory");
#endif
}
#endif // HAVE_ASYNC_LDS

__device__ __forceinline__ float geluf(float x) {
  return 0.5f * x * (1.0f + erff(x * 0.70710678118654752f));
}

__device__ __forceinline__ v8f zero_v8f() {
  v8f r;
#pragma unroll
  for (int i = 0; i < 8; ++i) r[i] = 0.0f;
  return r;
}

__device__ __forceinline__ v16bf pack16(v8bf lo, v8bf hi) {
  v16bf r;
#pragma unroll
  for (int i = 0; i < 8; ++i) { r[i] = lo[i]; r[i + 8] = hi[i]; }
  return r;
}

// ---------------------------------------------------------------------------
// Generic GEMM: C[M,N] = A[M,K] (bf16, row-major) @ B[K,N], where B is passed
// TRANSPOSED: Bt[N,K] bf16 row-major.  Batched via blockIdx.z with element
// strides.  EPI: 0 = f32 store, 2 = gelu -> bf16 store.
// Block = 256 threads (8 waves), tile 128x128x32, waves 4(M) x 2(N),
// each wave owns 32x64 = 2x4 wmma accumulators.
// Tiles are double-buffered in LDS; staged with GLOBAL_LOAD_ASYNC_TO_LDS_B128
// when available (ASYNCcnt/s_wait_asynccnt), else a synchronous VGPR path.
// Out-of-range rows clamp their source address (WMMA rows/cols are
// independent; the affected C rows/cols are discarded at the guarded store).
// ---------------------------------------------------------------------------
template <int EPI>
__global__ __launch_bounds__(256) void gemm_bt_k(
    const __bf16* __restrict__ A, size_t sAz, int lda,
    const __bf16* __restrict__ Bt, size_t sBz, int ldb,
    void* __restrict__ Cv, size_t sCz, int ldc,
    int M, int N, int K)
{
  __shared__ __align__(16) __bf16 As[2][128][40];   // 32 + 8 pad, x2 buffers
  __shared__ __align__(16) __bf16 Bs[2][128][40];

  const int tid  = threadIdx.x;
  const int lane = tid & 31;
  const int wid  = tid >> 5;
  const int wm   = wid & 3;                // wave M index 0..3
  const int wn   = wid >> 2;               // wave N index 0..1
  const int m0   = blockIdx.y * 128;
  const int n0   = blockIdx.x * 128;

  A  += (size_t)blockIdx.z * sAz;
  Bt += (size_t)blockIdx.z * sBz;
  float*  Cf = (float*)Cv  + (size_t)blockIdx.z * sCz;
  __bf16* Cb = (__bf16*)Cv + (size_t)blockIdx.z * sCz;

  v8f acc[2][4];
#pragma unroll
  for (int i = 0; i < 2; ++i)
#pragma unroll
    for (int j = 0; j < 4; ++j) acc[i][j] = zero_v8f();

  const int lrow  = tid >> 1;              // 0..127
  const int lhalf = (tid & 1) * 16;        // 0 or 16

  // clamped per-thread source rows (see note above)
  const int ra = (m0 + lrow < M) ? (m0 + lrow) : (M - 1);
  const int rb = (n0 + lrow < N) ? (n0 + lrow) : (N - 1);
  const __bf16* pa = A  + (size_t)ra * lda + lhalf;
  const __bf16* pb = Bt + (size_t)rb * ldb + lhalf;

  auto stage = [&](int buf, int k0) {
#if HAVE_ASYNC_LDS
    __builtin_amdgcn_global_load_async_to_lds_b128(
        (async_gptr)(pa + k0),     (async_lptr)&As[buf][lrow][lhalf],     0, 0);
    __builtin_amdgcn_global_load_async_to_lds_b128(
        (async_gptr)(pa + k0 + 8), (async_lptr)&As[buf][lrow][lhalf + 8], 0, 0);
    __builtin_amdgcn_global_load_async_to_lds_b128(
        (async_gptr)(pb + k0),     (async_lptr)&Bs[buf][lrow][lhalf],     0, 0);
    __builtin_amdgcn_global_load_async_to_lds_b128(
        (async_gptr)(pb + k0 + 8), (async_lptr)&Bs[buf][lrow][lhalf + 8], 0, 0);
#else
    v8bf va = *(const v8bf*)(pa + k0);
    v8bf vb = *(const v8bf*)(pa + k0 + 8);
    v8bf vc = *(const v8bf*)(pb + k0);
    v8bf vd = *(const v8bf*)(pb + k0 + 8);
    *(v8bf*)&As[buf][lrow][lhalf]     = va;
    *(v8bf*)&As[buf][lrow][lhalf + 8] = vb;
    *(v8bf*)&Bs[buf][lrow][lhalf]     = vc;
    *(v8bf*)&Bs[buf][lrow][lhalf + 8] = vd;
#endif
  };

  stage(0, 0);
  int buf = 0;
  for (int k0 = 0; k0 < K; k0 += 32) {
    const bool more = (k0 + 32) < K;
    if (more) stage(buf ^ 1, k0 + 32);
#if HAVE_ASYNC_LDS
    // 4 async instructions per wave per tile are outstanding for the
    // prefetched (next) tile; current tile is complete at count <= 4.
    if (more) wait_async4(); else wait_async0();
#endif
    __syncthreads();

    const int hh = lane >> 4;
    const int ml = lane & 15;
    v16bf af[2], bfb[4];
#pragma unroll
    for (int i = 0; i < 2; ++i) {
      int r = wm * 32 + i * 16 + ml;
      v8bf lo = *(const v8bf*)&As[buf][r][hh * 8];
      v8bf hi = *(const v8bf*)&As[buf][r][16 + hh * 8];
      af[i] = pack16(lo, hi);
    }
#pragma unroll
    for (int j = 0; j < 4; ++j) {
      int r = wn * 64 + j * 16 + ml;
      v8bf lo = *(const v8bf*)&Bs[buf][r][hh * 8];
      v8bf hi = *(const v8bf*)&Bs[buf][r][16 + hh * 8];
      bfb[j] = pack16(lo, hi);
    }
#pragma unroll
    for (int i = 0; i < 2; ++i)
#pragma unroll
      for (int j = 0; j < 4; ++j)
        acc[i][j] = __builtin_amdgcn_wmma_f32_16x16x32_bf16(
            false, af[i], false, bfb[j], (short)0, acc[i][j], false, false);
    __syncthreads();    // protect buf from next iteration's staging
    buf ^= 1;
  }

  // ---- epilogue / store ----
  const int hh = lane >> 4;
  const int nl = lane & 15;
#pragma unroll
  for (int i = 0; i < 2; ++i)
#pragma unroll
    for (int j = 0; j < 4; ++j) {
      int n = n0 + wn * 64 + j * 16 + nl;
      if (n >= N) continue;
#pragma unroll
      for (int r = 0; r < 8; ++r) {
        int m = m0 + wm * 32 + i * 16 + r + 8 * hh;
        if (m >= M) continue;
        float v = acc[i][j][r];
        if (EPI == 2) {
          Cb[(size_t)m * ldc + n] = (__bf16)geluf(v);
        } else {
          Cf[(size_t)m * ldc + n] = v;
        }
      }
    }
}

// ---------------------------------------------------------------------------
// MoE routed experts: gathered WMMA GEMMs (wave per 16x16 tile).
// ---------------------------------------------------------------------------
__global__ __launch_bounds__(32) void moe_up_gather_k(
    const __bf16* __restrict__ X,       // [BTOK, HDIM] ln2 output
    const __bf16* __restrict__ WuT,     // [NEXP, EDIM, HDIM]
    const int* __restrict__ tok, const int* __restrict__ cnt,
    __bf16* __restrict__ eup)           // [NEXP, CAP, EDIM]
{
  const int e = blockIdx.z;
  const int c = cnt[e];
  const int mt = blockIdx.y;
  if (mt * 16 >= c) return;
  const int nt = blockIdx.x;            // 0..31  (N = 512)
  const int lane = threadIdx.x;
  const int hh = lane >> 4, ml = lane & 15;

  int ri = mt * 16 + ml;
  int tk = (ri < c) ? tok[e * CAP + ri] : 0;
  const __bf16* arow = X + (size_t)tk * HDIM;
  const __bf16* brow = WuT + ((size_t)e * EDIM + (nt * 16 + ml)) * HDIM;

  v8f acc = zero_v8f();
#pragma unroll 4
  for (int k0 = 0; k0 < HDIM; k0 += 32) {
    __builtin_prefetch((const void*)(arow + k0 + 256), 0, 0);
    __builtin_prefetch((const void*)(brow + k0 + 256), 0, 0);
    v8bf alo = *(const v8bf*)(arow + k0 + hh * 8);
    v8bf ahi = *(const v8bf*)(arow + k0 + 16 + hh * 8);
    v8bf blo = *(const v8bf*)(brow + k0 + hh * 8);
    v8bf bhi = *(const v8bf*)(brow + k0 + 16 + hh * 8);
    acc = __builtin_amdgcn_wmma_f32_16x16x32_bf16(
        false, pack16(alo, ahi), false, pack16(blo, bhi), (short)0, acc, false, false);
  }
  int n = nt * 16 + ml;
#pragma unroll
  for (int r = 0; r < 8; ++r) {
    int row = mt * 16 + r + 8 * hh;
    if (row < c)
      eup[((size_t)e * CAP + row) * EDIM + n] = (__bf16)geluf(acc[r]);
  }
}

__global__ __launch_bounds__(32) void moe_down_gather_k(
    const __bf16* __restrict__ eup,     // [NEXP, CAP, EDIM]
    const __bf16* __restrict__ WdT,     // [NEXP, HDIM, EDIM]
    const int* __restrict__ tok, const float* __restrict__ gate,
    const int* __restrict__ cnt,
    float* __restrict__ routed)         // [BTOK, HDIM], atomically accumulated
{
  const int e = blockIdx.z;
  const int c = cnt[e];
  const int mt = blockIdx.y;
  if (mt * 16 >= c) return;
  const int nt = blockIdx.x;            // 0..63 (N = 1024)
  const int lane = threadIdx.x;
  const int hh = lane >> 4, ml = lane & 15;

  const __bf16* arow = eup + ((size_t)e * CAP + mt * 16 + ml) * EDIM;
  const __bf16* brow = WdT + ((size_t)e * HDIM + (nt * 16 + ml)) * EDIM;

  v8f acc = zero_v8f();
#pragma unroll 4
  for (int k0 = 0; k0 < EDIM; k0 += 32) {
    __builtin_prefetch((const void*)(arow + k0 + 256), 0, 0);
    __builtin_prefetch((const void*)(brow + k0 + 256), 0, 0);
    v8bf alo = *(const v8bf*)(arow + k0 + hh * 8);
    v8bf ahi = *(const v8bf*)(arow + k0 + 16 + hh * 8);
    v8bf blo = *(const v8bf*)(brow + k0 + hh * 8);
    v8bf bhi = *(const v8bf*)(brow + k0 + 16 + hh * 8);
    acc = __builtin_amdgcn_wmma_f32_16x16x32_bf16(
        false, pack16(alo, ahi), false, pack16(blo, bhi), (short)0, acc, false, false);
  }
  int n = nt * 16 + ml;
#pragma unroll
  for (int r = 0; r < 8; ++r) {
    int row = mt * 16 + r + 8 * hh;
    if (row < c) {
      int t = tok[e * CAP + row];
      float g = gate[e * CAP + row];
      atomicAdd(&routed[(size_t)t * HDIM + n], g * acc[r]);
    }
  }
}

// ---------------------------------------------------------------------------
// Router: one wave per token. softmax over 10 experts, top-3, bucket append.
// ---------------------------------------------------------------------------
__global__ __launch_bounds__(32) void router_k(
    const __bf16* __restrict__ x2, const float* __restrict__ Wr,
    int* __restrict__ cnt, int* __restrict__ tok, float* __restrict__ gate)
{
  const int t = blockIdx.x;
  const int lane = threadIdx.x;
  const __bf16* x = x2 + (size_t)t * HDIM;
  float acc[NEXP];
#pragma unroll
  for (int j = 0; j < NEXP; ++j) acc[j] = 0.0f;
  for (int k = lane; k < HDIM; k += 32) {
    float xv = (float)x[k];
    const float* wr = Wr + (size_t)k * NEXP;
#pragma unroll
    for (int j = 0; j < NEXP; ++j) acc[j] += xv * wr[j];
  }
#pragma unroll
  for (int j = 0; j < NEXP; ++j)
    for (int o = 16; o > 0; o >>= 1) acc[j] += __shfl_xor(acc[j], o, 32);

  if (lane == 0) {
    float mx = acc[0];
#pragma unroll
    for (int j = 1; j < NEXP; ++j) mx = fmaxf(mx, acc[j]);
    float p[NEXP], ssum = 0.0f;
#pragma unroll
    for (int j = 0; j < NEXP; ++j) { p[j] = __expf(acc[j] - mx); ssum += p[j]; }
    float inv = 1.0f / ssum;
#pragma unroll
    for (int j = 0; j < NEXP; ++j) p[j] *= inv;
    for (int k = 0; k < KTOP; ++k) {
      int bi = 0; float bv = p[0];
#pragma unroll
      for (int j = 1; j < NEXP; ++j) if (p[j] > bv) { bv = p[j]; bi = j; }
      int pos = atomicAdd(&cnt[bi], 1);
      tok[bi * CAP + pos]  = t;
      gate[bi * CAP + pos] = bv;
      p[bi] = -1.0f;
    }
  }
}

// ---------------------------------------------------------------------------
// LayerNorm over H=1024; one block per row; OUT = __bf16 or float.
// row = blockIdx.x * rowmul + rowoff ; output row = blockIdx.x.
// ---------------------------------------------------------------------------
template <typename OUT>
__global__ __launch_bounds__(256) void layernorm_k(
    const float* __restrict__ in, const float* __restrict__ g,
    const float* __restrict__ b, OUT* __restrict__ out,
    int rowmul, int rowoff)
{
  __shared__ float sw[8], sw2[8];
  const int row = blockIdx.x * rowmul + rowoff;
  const float* x = in + (size_t)row * HDIM;
  float v[4], s = 0.0f, s2 = 0.0f;
#pragma unroll
  for (int i = 0; i < 4; ++i) {
    v[i] = x[threadIdx.x + i * 256];
    s += v[i]; s2 += v[i] * v[i];
  }
  for (int o = 16; o > 0; o >>= 1) { s += __shfl_xor(s, o, 32); s2 += __shfl_xor(s2, o, 32); }
  int wid = threadIdx.x >> 5;
  if ((threadIdx.x & 31) == 0) { sw[wid] = s; sw2[wid] = s2; }
  __syncthreads();
  if (threadIdx.x < 32) {
    float a  = (threadIdx.x < 8) ? sw[threadIdx.x]  : 0.0f;
    float a2 = (threadIdx.x < 8) ? sw2[threadIdx.x] : 0.0f;
    for (int o = 4; o > 0; o >>= 1) { a += __shfl_xor(a, o, 32); a2 += __shfl_xor(a2, o, 32); }
    if (threadIdx.x == 0) { sw[0] = a; sw2[0] = a2; }
  }
  __syncthreads();
  float mean = sw[0] * (1.0f / HDIM);
  float var  = sw2[0] * (1.0f / HDIM) - mean * mean;
  float inv  = rsqrtf(var + 1e-5f);
  OUT* y = out + (size_t)blockIdx.x * HDIM;
#pragma unroll
  for (int i = 0; i < 4; ++i) {
    int c = threadIdx.x + i * 256;
    y[c] = (OUT)((v[i] - mean) * inv * g[c] + b[c]);
  }
}

// ---------------------------------------------------------------------------
// Causal softmax row kernel: scores[h*T*T + t*T + j] (f32) -> att bf16.
// ---------------------------------------------------------------------------
__global__ __launch_bounds__(256) void softmax_causal_k(
    const float* __restrict__ scores, __bf16* __restrict__ att)
{
  const int hq = blockIdx.x;            // h*1024 + t
  const int t = hq & 1023;
  const float* s = scores + (size_t)hq * TSEQ;
  __bf16* ob = att + (size_t)hq * TSEQ;
  __shared__ float sm[8], sm2[8];
  float v[4], mx = -3.0e38f;
#pragma unroll
  for (int i = 0; i < 4; ++i) {
    int j = threadIdx.x + i * 256;
    float x = s[j] * 0.125f;            // 1/sqrt(64)
    if (j > t) x = -3.0e38f;
    v[i] = x; mx = fmaxf(mx, x);
  }
  for (int o = 16; o > 0; o >>= 1) mx = fmaxf(mx, __shfl_xor(mx, o, 32));
  int wid = threadIdx.x >> 5;
  if ((threadIdx.x & 31) == 0) sm[wid] = mx;
  __syncthreads();
  if (threadIdx.x < 32) {
    float a = (threadIdx.x < 8) ? sm[threadIdx.x] : -3.0e38f;
    for (int o = 4; o > 0; o >>= 1) a = fmaxf(a, __shfl_xor(a, o, 32));
    if (threadIdx.x == 0) sm[0] = a;
  }
  __syncthreads();
  mx = sm[0];
  float ssum = 0.0f;
#pragma unroll
  for (int i = 0; i < 4; ++i) { v[i] = __expf(v[i] - mx); ssum += v[i]; }
  for (int o = 16; o > 0; o >>= 1) ssum += __shfl_xor(ssum, o, 32);
  if ((threadIdx.x & 31) == 0) sm2[wid] = ssum;
  __syncthreads();
  if (threadIdx.x < 32) {
    float a = (threadIdx.x < 8) ? sm2[threadIdx.x] : 0.0f;
    for (int o = 4; o > 0; o >>= 1) a += __shfl_xor(a, o, 32);
    if (threadIdx.x == 0) sm2[0] = a;
  }
  __syncthreads();
  float inv = 1.0f / sm2[0];
#pragma unroll
  for (int i = 0; i < 4; ++i)
    ob[threadIdx.x + i * 256] = (__bf16)(v[i] * inv);
}

// ---------------------------------------------------------------------------
// Small elementwise / data-movement kernels.
// ---------------------------------------------------------------------------
__global__ void embed_k(const int* __restrict__ ids, const float* __restrict__ Wemb,
                        const float* __restrict__ Wpos, float* __restrict__ h)
{
  size_t i = (size_t)blockIdx.x * 256 + threadIdx.x;   // over BTOK*HDIM
  int row = (int)(i >> 10);
  int c   = (int)(i & 1023);
  int t   = row & 1023;
  int id  = ids[row];
  h[i] = Wemb[(size_t)id * HDIM + c] + Wpos[(size_t)t * HDIM + c];
}

__global__ void split_qkv_k(const float* __restrict__ qkv,
                            __bf16* __restrict__ qh, __bf16* __restrict__ kh,
                            __bf16* __restrict__ vt)
{
  size_t i = (size_t)blockIdx.x * 256 + threadIdx.x;   // over BTOK*HDIM
  int row = (int)(i >> 10);          // b*T + t
  int c   = (int)(i & 1023);
  int b   = row >> 10;
  int t   = row & 1023;
  int hh  = c >> 6;
  int d   = c & 63;
  const float* q = qkv + (size_t)row * 3072;
  size_t base = ((size_t)(b * NHEAD + hh) * TSEQ + t) * DHEAD + d;
  qh[base] = (__bf16)q[c];
  kh[base] = (__bf16)q[1024 + c];
  vt[((size_t)(b * NHEAD + hh) * DHEAD + d) * TSEQ + t] = (__bf16)q[2048 + c];
}

__global__ void f2bf_k(const float* __restrict__ in, __bf16* __restrict__ out, size_t n)
{
  size_t i = (size_t)blockIdx.x * 256 + threadIdx.x;
  if (i < n) out[i] = (__bf16)in[i];
}

__global__ void add2_k(const float* __restrict__ a, const float* __restrict__ b,
                       float* __restrict__ out, size_t n)
{
  size_t i = (size_t)blockIdx.x * 256 + threadIdx.x;
  if (i < n) out[i] = a[i] + b[i];
}

__global__ void add3_inplace_k(float* __restrict__ h, const float* __restrict__ s,
                               const float* __restrict__ r, size_t n)
{
  size_t i = (size_t)blockIdx.x * 256 + threadIdx.x;
  if (i < n) h[i] = h[i] + s[i] + r[i];
}

__global__ void zero_f32_k(float* __restrict__ p, size_t n)
{
  size_t i = (size_t)blockIdx.x * 256 + threadIdx.x;
  if (i < n) p[i] = 0.0f;
}

__global__ void zero_i32_k(int* __restrict__ p, int n)
{
  int i = (int)threadIdx.x;
  if (i < n) p[i] = 0;
}

// Batched transpose + f32->bf16: in [z][rows][cols] -> out [z][cols][rows]
__global__ void transpose_bf_k(const float* __restrict__ in, __bf16* __restrict__ out,
                               int rows, int cols)
{
  size_t nrc = (size_t)rows * cols;
  const float* ib = in + (size_t)blockIdx.z * nrc;
  __bf16* ob = out + (size_t)blockIdx.z * nrc;
  size_t i = (size_t)blockIdx.x * 256 + threadIdx.x;
  if (i < nrc) {
    int r = (int)(i / cols), c = (int)(i % cols);
    ob[(size_t)c * rows + r] = (__bf16)ib[i];
  }
}

// Tied head: logits[b, v] = dot(hln[b], Wemb[v]) ; one wave per vocab row.
__global__ __launch_bounds__(256) void head_k(
    const float* __restrict__ hln, const float* __restrict__ Wemb,
    float* __restrict__ out)
{
  int wid = threadIdx.x >> 5, lane = threadIdx.x & 31;
  int v = blockIdx.x * 8 + wid;
  const float* wr = Wemb + (size_t)v * HDIM;
  float a0 = 0, a1 = 0, a2 = 0, a3 = 0;
  for (int k = lane; k < HDIM; k += 32) {
    float w = wr[k];
    a0 += hln[k] * w;
    a1 += hln[1024 + k] * w;
    a2 += hln[2048 + k] * w;
    a3 += hln[3072 + k] * w;
  }
  for (int o = 16; o > 0; o >>= 1) {
    a0 += __shfl_xor(a0, o, 32);
    a1 += __shfl_xor(a1, o, 32);
    a2 += __shfl_xor(a2, o, 32);
    a3 += __shfl_xor(a3, o, 32);
  }
  if (lane == 0) {
    out[v]             = a0;
    out[VOCAB + v]     = a1;
    out[2 * VOCAB + v] = a2;
    out[3 * VOCAB + v] = a3;
  }
}

// ---------------------------------------------------------------------------
// Host-side orchestration.
// ---------------------------------------------------------------------------
extern "C" void kernel_launch(void* const* d_in, const int* in_sizes, int n_in,
                              void* d_out, int out_size, void* d_ws, size_t ws_size,
                              hipStream_t stream)
{
  (void)in_sizes; (void)n_in; (void)out_size; (void)ws_size;

  const int*   x_ids = (const int*)  d_in[0];
  const float* Wemb  = (const float*)d_in[1];
  const float* Wpos  = (const float*)d_in[2];
  const float* ln1g  = (const float*)d_in[3];
  const float* ln1b  = (const float*)d_in[4];
  const float* Wqkv  = (const float*)d_in[5];
  const float* Wo    = (const float*)d_in[6];
  const float* ln2g  = (const float*)d_in[7];
  const float* ln2b  = (const float*)d_in[8];
  const float* Wr    = (const float*)d_in[9];
  const float* Wsu   = (const float*)d_in[10];
  const float* Wsd   = (const float*)d_in[11];
  const float* Wu    = (const float*)d_in[12];
  const float* Wd    = (const float*)d_in[13];
  const float* lnfg  = (const float*)d_in[14];
  const float* lnfb  = (const float*)d_in[15];

  // ---- workspace carve-up ----
  char* base = (char*)d_ws;
  size_t off = 0;
  auto alloc = [&](size_t bytes) -> void* {
    off = (off + 255) & ~(size_t)255;
    void* p = base + off;
    off += bytes;
    return p;
  };
  const size_t BH = (size_t)BTOK * HDIM;

  float*  h        = (float*) alloc(BH * 4);
  float*  a_res    = (float*) alloc(BH * 4);
  __bf16* x1       = (__bf16*)alloc(BH * 2);
  __bf16* x2       = (__bf16*)alloc(BH * 2);
  float*  qkv      = (float*) alloc((size_t)BTOK * 3072 * 4);
  __bf16* qh       = (__bf16*)alloc(BH * 2);
  __bf16* kh       = (__bf16*)alloc(BH * 2);
  __bf16* vt       = (__bf16*)alloc(BH * 2);
  float*  scores   = (float*) alloc((size_t)NHEAD * TSEQ * TSEQ * 4);   // 64 MB
  __bf16* att      = (__bf16*)alloc((size_t)NHEAD * TSEQ * TSEQ * 2);   // 32 MB
  float*  attnout  = (float*) alloc(BH * 4);
  __bf16* ybf      = (__bf16*)alloc(BH * 2);
  float*  proj     = (float*) alloc(BH * 4);
  __bf16* supb     = (__bf16*)alloc((size_t)BTOK * EDIM * 2);
  float*  sharedf  = (float*) alloc(BH * 4);
  float*  routed   = (float*) alloc(BH * 4);
  __bf16* eup      = (__bf16*)alloc((size_t)NEXP * CAP * EDIM * 2);
  int*    tok      = (int*)   alloc((size_t)NEXP * CAP * 4);
  float*  gate     = (float*) alloc((size_t)NEXP * CAP * 4);
  int*    cnt      = (int*)   alloc(64);
  float*  hln      = (float*) alloc((size_t)4 * HDIM * 4);
  __bf16* wqkvT    = (__bf16*)alloc((size_t)NLAYER * 3072 * HDIM * 2);
  __bf16* woT      = (__bf16*)alloc((size_t)NLAYER * HDIM * HDIM * 2);
  __bf16* wsuT     = (__bf16*)alloc((size_t)NLAYER * EDIM * HDIM * 2);
  __bf16* wsdT     = (__bf16*)alloc((size_t)NLAYER * HDIM * EDIM * 2);
  __bf16* wuT      = (__bf16*)alloc((size_t)NLAYER * NEXP * EDIM * HDIM * 2);
  __bf16* wdT      = (__bf16*)alloc((size_t)NLAYER * NEXP * HDIM * EDIM * 2);

  // ---- weight conversion (transpose + bf16) ----
  transpose_bf_k<<<dim3(12288, 1, NLAYER), 256, 0, stream>>>(Wqkv, wqkvT, HDIM, 3072);
  transpose_bf_k<<<dim3(4096, 1, NLAYER), 256, 0, stream>>>(Wo,   woT,   HDIM, HDIM);
  transpose_bf_k<<<dim3(2048, 1, NLAYER), 256, 0, stream>>>(Wsu,  wsuT,  HDIM, EDIM);
  transpose_bf_k<<<dim3(2048, 1, NLAYER), 256, 0, stream>>>(Wsd,  wsdT,  EDIM, HDIM);
  transpose_bf_k<<<dim3(2048, 1, NLAYER * NEXP), 256, 0, stream>>>(Wu, wuT, HDIM, EDIM);
  transpose_bf_k<<<dim3(2048, 1, NLAYER * NEXP), 256, 0, stream>>>(Wd, wdT, EDIM, HDIM);

  // ---- embedding ----
  embed_k<<<16384, 256, 0, stream>>>(x_ids, Wemb, Wpos, h);

  for (int l = 0; l < NLAYER; ++l) {
    const __bf16* wqkvT_l = wqkvT + (size_t)l * 3072 * HDIM;
    const __bf16* woT_l   = woT   + (size_t)l * HDIM * HDIM;
    const __bf16* wsuT_l  = wsuT  + (size_t)l * EDIM * HDIM;
    const __bf16* wsdT_l  = wsdT  + (size_t)l * HDIM * EDIM;
    const __bf16* wuT_l   = wuT   + (size_t)l * NEXP * EDIM * HDIM;
    const __bf16* wdT_l   = wdT   + (size_t)l * NEXP * HDIM * EDIM;

    // LN1 -> x1 (bf16)
    layernorm_k<__bf16><<<BTOK, 256, 0, stream>>>(h, ln1g + l * HDIM, ln1b + l * HDIM, x1, 1, 0);

    // QKV GEMM: [4096,3072] = x1[4096,1024] @ Wqkv
    gemm_bt_k<0><<<dim3(24, 32, 1), 256, 0, stream>>>(
        x1, 0, HDIM, wqkvT_l, 0, HDIM, qkv, 0, 3072, BTOK, 3072, HDIM);

    split_qkv_k<<<16384, 256, 0, stream>>>(qkv, qh, kh, vt);

    for (int b = 0; b < 4; ++b) {
      const __bf16* qb = qh + (size_t)b * NHEAD * TSEQ * DHEAD;
      const __bf16* kb = kh + (size_t)b * NHEAD * TSEQ * DHEAD;
      const __bf16* vb = vt + (size_t)b * NHEAD * DHEAD * TSEQ;
      // scores[h] = Q[h] @ K[h]^T   (Bt = K, already [N=T, K=64] row-major)
      gemm_bt_k<0><<<dim3(8, 8, NHEAD), 256, 0, stream>>>(
          qb, (size_t)TSEQ * DHEAD, DHEAD,
          kb, (size_t)TSEQ * DHEAD, DHEAD,
          scores, (size_t)TSEQ * TSEQ, TSEQ, TSEQ, TSEQ, DHEAD);
      // mask + softmax -> att bf16
      softmax_causal_k<<<NHEAD * TSEQ, 256, 0, stream>>>(scores, att);
      // y[h] = att[h] @ V[h]   (Bt = V^T, [N=64, K=T] row-major)
      gemm_bt_k<0><<<dim3(1, 8, NHEAD), 256, 0, stream>>>(
          att, (size_t)TSEQ * TSEQ, TSEQ,
          vb, (size_t)DHEAD * TSEQ, TSEQ,
          attnout + (size_t)b * TSEQ * HDIM, (size_t)DHEAD, HDIM,
          TSEQ, DHEAD, TSEQ);
    }

    // out projection + residual
    f2bf_k<<<16384, 256, 0, stream>>>(attnout, ybf, BH);
    gemm_bt_k<0><<<dim3(8, 32, 1), 256, 0, stream>>>(
        ybf, 0, HDIM, woT_l, 0, HDIM, proj, 0, HDIM, BTOK, HDIM, HDIM);
    add2_k<<<16384, 256, 0, stream>>>(h, proj, a_res, BH);

    // LN2 -> x2 (bf16)
    layernorm_k<__bf16><<<BTOK, 256, 0, stream>>>(a_res, ln2g + l * HDIM, ln2b + l * HDIM, x2, 1, 0);

    // shared expert: gelu(x2 @ Wsu) @ Wsd
    gemm_bt_k<2><<<dim3(4, 32, 1), 256, 0, stream>>>(
        x2, 0, HDIM, wsuT_l, 0, HDIM, supb, 0, EDIM, BTOK, EDIM, HDIM);
    gemm_bt_k<0><<<dim3(8, 32, 1), 256, 0, stream>>>(
        supb, 0, EDIM, wsdT_l, 0, EDIM, sharedf, 0, HDIM, BTOK, HDIM, EDIM);

    // routed experts: top-3 gather
    zero_f32_k<<<16384, 256, 0, stream>>>(routed, BH);
    zero_i32_k<<<1, 32, 0, stream>>>(cnt, NEXP);
    router_k<<<BTOK, 32, 0, stream>>>(x2, Wr + (size_t)l * HDIM * NEXP, cnt, tok, gate);
    moe_up_gather_k<<<dim3(EDIM / 16, CAP / 16, NEXP), 32, 0, stream>>>(
        x2, wuT_l, tok, cnt, eup);
    moe_down_gather_k<<<dim3(HDIM / 16, CAP / 16, NEXP), 32, 0, stream>>>(
        eup, wdT_l, tok, gate, cnt, routed);

    // h = (shared + routed) + h   (residual from BLOCK INPUT, per reference)
    add3_inplace_k<<<16384, 256, 0, stream>>>(h, sharedf, routed, BH);
  }

  // final LN on the last position of each batch, then tied head
  layernorm_k<float><<<4, 256, 0, stream>>>(h, lnfg, lnfb, hln, TSEQ, TSEQ - 1);
  head_k<<<VOCAB / 8, 256, 0, stream>>>(hln, Wemb, (float*)d_out);
}
